// GATNet_10943576670986
// MI455X (gfx1250) — matrix operand order repaired and checked by reading, not compiled, not run on hardware
//
#include <hip/hip_runtime.h>
#include <hip/hip_bf16.h>
#include <math.h>

// ---------------------------------------------------------------------------
// GATv2 (4 layers, H=4, D=128, HD=512) + BN + pool + MLP for gfx1250 (wave32)
// GEMMs on v_wmma_f32_16x16x32_f16 with fragment-packed B (contiguous 32B per
// lane -> global_load_b128) and LDS-staged A (ds_load_b128). Edge phase is
// gather/atomic bound and sized for the 192MB L2 (xl+xr ~= 205MB f32).
// ---------------------------------------------------------------------------

typedef __attribute__((ext_vector_type(16))) _Float16 v16h;
typedef __attribute__((ext_vector_type(8)))  _Float16 v8h;
typedef __attribute__((ext_vector_type(8)))  float    v8f;

#define HD    512
#define HEADS 4
#define DHEAD 128
#define BN_EPS 1e-5f

// ---------------- helpers --------------------------------------------------

__device__ __forceinline__ unsigned f32_enc(float f) {
  unsigned u = __float_as_uint(f);
  return (u & 0x80000000u) ? ~u : (u | 0x80000000u);
}
__device__ __forceinline__ float f32_dec(unsigned x) {
  unsigned u = (x & 0x80000000u) ? (x & 0x7fffffffu) : ~x;
  return __uint_as_float(u);
}

__global__ void k_fill_f32(float* p, float v, long n) {
  long i = (long)blockIdx.x * blockDim.x + threadIdx.x;
  long step = (long)gridDim.x * blockDim.x;
  for (; i < n; i += step) p[i] = v;
}

__global__ void k_fill_u32(unsigned* p, unsigned v, long n) {
  long i = (long)blockIdx.x * blockDim.x + threadIdx.x;
  long step = (long)gridDim.x * blockDim.x;
  for (; i < n; i += step) p[i] = v;
}

// acc[n*HD + c] = bias[c]   (segment_sum init: + bias pre-loaded)
__global__ void k_init_bias(float* acc, const float* __restrict__ bias, long n) {
  long i = (long)blockIdx.x * blockDim.x + threadIdx.x;
  long step = (long)gridDim.x * blockDim.x;
  for (; i < n; i += step) acc[i] = bias[i & (HD - 1)];
}

// ---------------- embedding (identity tables with max_norm) ----------------

__global__ void k_embed(const int* __restrict__ x,
                        const float* t0, const float* t1, const float* t2,
                        const float* t3, const float* t4, const float* t5,
                        const float* t6, const float* t7,
                        _Float16* __restrict__ h, int N) {
  int n = blockIdx.x * blockDim.x + threadIdx.x;
  if (n >= N) return;
  const float* T[8] = {t0, t1, t2, t3, t4, t5, t6, t7};
  const int lens[8] = {44, 7, 6, 7, 2, 2, 6, 6};
  _Float16* row = h + (size_t)n * 96;
  for (int i = 0; i < 96; ++i) row[i] = (_Float16)0.f;   // zero-pad K: 80 -> 96
  int off = 0;
  for (int f = 0; f < 8; ++f) {
    int L = lens[f];
    int r = x[n * 8 + f];
    const float* tr = T[f] + (size_t)r * L;
    float nrm = 0.f;
    for (int j = 0; j < L; ++j) nrm += tr[j] * tr[j];
    nrm = sqrtf(nrm);
    float sc = fminf(1.f, 1.f / (nrm + 1e-7f));          // torch max_norm=1.0
    for (int j = 0; j < L; ++j) row[off + j] = (_Float16)(tr[j] * sc);
    off += L;
  }
}

// ---------------- weight repack into WMMA-fragment order -------------------
// Bp flat index = ((kb*ntiles + nt)*32 + lane)*16 + e, where the 16 halves a
// lane needs for tile nt / k-block kb are contiguous (32B -> global_load_b128).
// lane n = nt*16 + (lane&15); K offset = kb*32 + ((lane>>4)<<3) + (e&7) + ((e>>3)<<4).

__global__ void k_b_pack(const float* __restrict__ W, _Float16* __restrict__ Bp,
                         int Ksrc, int Kpad, int Nc) {
  long i = (long)blockIdx.x * blockDim.x + threadIdx.x;
  long tot = (long)Kpad * Nc;
  long step = (long)gridDim.x * blockDim.x;
  int ntiles = Nc >> 4;
  for (; i < tot; i += step) {
    int e  = (int)(i & 15);
    int l  = (int)((i >> 4) & 31);
    long t = i >> 9;
    int nt = (int)(t % ntiles);
    int kb = (int)(t / ntiles);
    int n  = nt * 16 + (l & 15);
    int k  = kb * 32 + ((l >> 4) << 3) + (e & 7) + ((e >> 3) << 4);
    Bp[i] = (k < Ksrc) ? (_Float16)W[(size_t)k * Nc + n] : (_Float16)0.f;
  }
}

// ---------------- WMMA GEMM: C[MxNc] f32 = A[MxK] f16 @ B[KxNc] f16 --------
// 128 threads = 4 waves; each wave owns TWO adjacent 16x16 D tiles (A reuse),
// block tile = 16 x 128. A strip (16 x K <= 16KB) staged through LDS.

template <int K>
__global__ __launch_bounds__(128)
void k_gemm_f16_wmma(const _Float16* __restrict__ A, const _Float16* __restrict__ Bp,
                     float* __restrict__ C, int M, int Nc) {
  __shared__ _Float16 sA[16 * K];
  const int m0 = blockIdx.y * 16;
  const bool full = (m0 + 16 <= M);
  for (int idx = threadIdx.x; idx < 16 * K; idx += 128) {
    int r = idx / K, cc = idx - r * K;
    int m = m0 + r;
    sA[idx] = (full || m < M) ? A[(size_t)m * K + cc] : (_Float16)0.f;
  }
  __syncthreads();

  const int lane   = threadIdx.x & 31;
  const int wv     = threadIdx.x >> 5;
  const int ntiles = Nc >> 4;
  const int nt0    = (blockIdx.x * 128 + wv * 32) >> 4;   // first of 2 tiles
  const int kbA    = (lane >> 4) << 3;                    // 0 / 8
  const _Float16* Ar = sA + (size_t)(lane & 15) * K;

  v8f acc0 = {}, acc1 = {};
#pragma unroll 4
  for (int kb = 0; kb < K / 32; ++kb) {
    const int k0 = kb * 32;
    // A fragment: two contiguous 16B LDS chunks (ISA 16-bit A 16x32 layout)
    v8h lo = *(const v8h*)(Ar + k0 + kbA);
    v8h hi = *(const v8h*)(Ar + k0 + kbA + 16);
    v16h a;
#pragma unroll
    for (int i = 0; i < 8; ++i) { a[i] = lo[i]; a[8 + i] = hi[i]; }
    // B fragments: contiguous 32B per lane from packed weights
    const _Float16* bbase = Bp + (((size_t)kb * ntiles + nt0) * 32 + lane) * 16;
    v16h b0 = *(const v16h*)bbase;
    v16h b1 = *(const v16h*)(bbase + 32 * 16);
    acc0 = __builtin_amdgcn_wmma_f32_16x16x32_f16(false, a, false, b0,
                                                  (short)0, acc0, false, false);
    acc1 = __builtin_amdgcn_wmma_f32_16x16x32_f16(false, a, false, b1,
                                                  (short)0, acc1, false, false);
  }

  const int mrow = m0 + ((lane >> 4) << 3);
  const int n0   = nt0 * 16 + (lane & 15);
  if (full) {
#pragma unroll
    for (int v = 0; v < 8; ++v) {
      C[(size_t)(mrow + v) * Nc + n0]      = acc0[v];
      C[(size_t)(mrow + v) * Nc + n0 + 16] = acc1[v];
    }
  } else {
#pragma unroll
    for (int v = 0; v < 8; ++v)
      if (mrow + v < M) {
        C[(size_t)(mrow + v) * Nc + n0]      = acc0[v];
        C[(size_t)(mrow + v) * Nc + n0 + 16] = acc1[v];
      }
  }
}

// ---------------- edge attention phase -------------------------------------
// wave per edge: lane handles 16 consecutive channels (never crosses a head);
// ef computed on the fly from edge_attr and We rows (avoids a 1.6GB ef tensor).

__global__ __launch_bounds__(256)
void k_edge_logit(const float* __restrict__ xl, const float* __restrict__ xr,
                  const float* __restrict__ eattr, const int* __restrict__ eidx,
                  const float* __restrict__ We, const float* __restrict__ att,
                  float* __restrict__ logit, unsigned* __restrict__ umax, int E) {
  int e = blockIdx.x * 8 + (threadIdx.x >> 5);
  if (e >= E) return;
  const int lane = threadIdx.x & 31;
  const int s = eidx[e], d = eidx[E + e];
  const float e0 = eattr[2 * e], e1 = eattr[2 * e + 1];
  const float* xls = xl + (size_t)s * HD;
  const float* xrd = xr + (size_t)d * HD;
  __builtin_prefetch(xls, 0, 1);   // -> global_prefetch_b8
  float acc = 0.f;
  const int c0 = lane * 16;
#pragma unroll 4
  for (int j = 0; j < 16; ++j) {
    int c = c0 + j;
    float sv = xls[c] + xrd[c] + e0 * We[c] + e1 * We[HD + c];
    sv = sv > 0.f ? sv : 0.2f * sv;            // GATv2 leaky slope
    acc += sv * att[c];                        // att flat: head = c/128
  }
  acc += __shfl_xor(acc, 1, 32);
  acc += __shfl_xor(acc, 2, 32);
  acc += __shfl_xor(acc, 4, 32);               // reduce 8-lane head group
  if ((lane & 7) == 0) {
    int h = lane >> 3;
    logit[(size_t)e * HEADS + h] = acc;
    atomicMax(&umax[(size_t)d * HEADS + h], f32_enc(acc));
  }
}

__global__ void k_edge_exp(const float* __restrict__ logit, const int* __restrict__ eidx,
                           const unsigned* __restrict__ umax, float* __restrict__ ex,
                           float* __restrict__ den, int E) {
  long i = (long)blockIdx.x * blockDim.x + threadIdx.x;
  long tot = (long)E * HEADS;
  long step = (long)gridDim.x * blockDim.x;
  for (; i < tot; i += step) {
    int e = (int)(i >> 2), h = (int)(i & 3);
    int d = eidx[E + e];
    float m = f32_dec(umax[(size_t)d * HEADS + h]);
    float v = expf(logit[i] - m);
    ex[i] = v;
    atomicAdd(&den[(size_t)d * HEADS + h], v);
  }
}

__global__ __launch_bounds__(256)
void k_edge_accum(const float* __restrict__ xl, const int* __restrict__ eidx,
                  const float* __restrict__ ex, const float* __restrict__ den,
                  float* __restrict__ acc, int E) {
  int e = blockIdx.x * 8 + (threadIdx.x >> 5);
  if (e >= E) return;
  const int lane = threadIdx.x & 31;
  const int s = eidx[e], d = eidx[E + e];
  const int h = lane >> 3;
  const float alpha = ex[(size_t)e * HEADS + h] /
                      (den[(size_t)d * HEADS + h] + 1e-16f);
  const float* xls = xl + (size_t)s * HD;
  float* out = acc + (size_t)d * HD;
  const int c0 = lane * 16;
#pragma unroll 4
  for (int j = 0; j < 16; ++j) {
    int c = c0 + j;
    atomicAdd(&out[c], alpha * xls[c]);
  }
}

// ---------------- BatchNorm / activations ----------------------------------

__global__ __launch_bounds__(256)
void k_bn_stats(const float* __restrict__ X, int rows, int cols,
                float* __restrict__ mu, float* __restrict__ rs) {
  __shared__ float s1[256], s2[256];
  const int c = blockIdx.x;
  const int tid = threadIdx.x;
  float a = 0.f, b = 0.f;
  for (int r = tid; r < rows; r += blockDim.x) {
    float v = X[(size_t)r * cols + c];
    a += v; b += v * v;
  }
  s1[tid] = a; s2[tid] = b;
  __syncthreads();
  for (int off = 128; off > 0; off >>= 1) {
    if (tid < off) { s1[tid] += s1[tid + off]; s2[tid] += s2[tid + off]; }
    __syncthreads();
  }
  if (tid == 0) {
    float m = s1[0] / rows;
    float v = s2[0] / rows - m * m;
    mu[c] = m;
    rs[c] = rsqrtf(v + BN_EPS);
  }
}

// BN + leaky(0.01); writes f32 (for pooling) and f16 (next layer GEMM input)
__global__ void k_bn_leaky(float* __restrict__ acc, const float* __restrict__ mu,
                           const float* __restrict__ rs, const float* __restrict__ g,
                           const float* __restrict__ b, _Float16* __restrict__ hf16,
                           long n) {
  long i = (long)blockIdx.x * blockDim.x + threadIdx.x;
  long step = (long)gridDim.x * blockDim.x;
  for (; i < n; i += step) {
    int c = (int)(i & (HD - 1));
    float v = g[c] * (acc[i] - mu[c]) * rs[c] + b[c];
    v = v > 0.f ? v : 0.01f * v;
    acc[i] = v;
    hf16[i] = (_Float16)v;
  }
}

// BN + relu (MLP), in place
__global__ void k_bn_relu(float* __restrict__ z, const float* __restrict__ mu,
                          const float* __restrict__ rs, const float* __restrict__ g,
                          const float* __restrict__ b, int rows, int cols) {
  long i = (long)blockIdx.x * blockDim.x + threadIdx.x;
  long tot = (long)rows * cols;
  long step = (long)gridDim.x * blockDim.x;
  for (; i < tot; i += step) {
    int c = (int)(i % cols);
    float v = g[c] * (z[i] - mu[c]) * rs[c] + b[c];
    z[i] = v > 0.f ? v : 0.f;
  }
}

// ---------------- pooling + MLP --------------------------------------------

__global__ void k_pool(const float* __restrict__ acc, const int* __restrict__ batch,
                       float* __restrict__ pooled, long n) {
  long i = (long)blockIdx.x * blockDim.x + threadIdx.x;
  long step = (long)gridDim.x * blockDim.x;
  for (; i < n; i += step) {
    int nd = (int)(i >> 9);
    int c  = (int)(i & (HD - 1));
    atomicAdd(&pooled[(size_t)batch[nd] * HD + c], acc[i]);
  }
}

__global__ void k_lin(const float* __restrict__ A, const float* __restrict__ W,
                      const float* __restrict__ bias, float* __restrict__ C,
                      int M, int K, int Nc) {
  long i = (long)blockIdx.x * blockDim.x + threadIdx.x;
  long tot = (long)M * Nc;
  long step = (long)gridDim.x * blockDim.x;
  for (; i < tot; i += step) {
    int m = (int)(i / Nc), n = (int)(i - (long)m * Nc);
    float s = bias[n];
    for (int k = 0; k < K; ++k) s += A[(size_t)m * K + k] * W[(size_t)k * Nc + n];
    C[i] = s;
  }
}

// ---------------- host orchestration ---------------------------------------

extern "C" void kernel_launch(void* const* d_in, const int* in_sizes, int n_in,
                              void* d_out, int out_size, void* d_ws, size_t ws_size,
                              hipStream_t stream) {
  const int N = in_sizes[0] / 8;        // 50000
  const int E = in_sizes[2] / 2;        // 800000
  const int G = out_size / 2;           // 512

  const int*   x     = (const int*)d_in[0];
  const int*   eidx  = (const int*)d_in[1];   // [2,E] row-major: src, dst
  const float* eattr = (const float*)d_in[2]; // [E,2]
  const int*   batch = (const int*)d_in[3];

  const float* emb[8];
  for (int i = 0; i < 8; ++i) emb[i] = (const float*)d_in[4 + i];

  struct Layer { const float *Wl, *Wr, *We, *att, *bias, *g, *b; } L[4];
  for (int l = 0; l < 4; ++l) {
    int base = 12 + 7 * l;                    // dict insertion order
    L[l].Wl   = (const float*)d_in[base + 0];
    L[l].Wr   = (const float*)d_in[base + 1];
    L[l].We   = (const float*)d_in[base + 2];
    L[l].att  = (const float*)d_in[base + 3];
    L[l].bias = (const float*)d_in[base + 4];
    L[l].g    = (const float*)d_in[base + 5];
    L[l].b    = (const float*)d_in[base + 6];
  }
  const float* W1  = (const float*)d_in[40];
  const float* b1  = (const float*)d_in[41];
  const float* g1  = (const float*)d_in[42];
  const float* bb1 = (const float*)d_in[43];
  const float* W2  = (const float*)d_in[44];
  const float* b2  = (const float*)d_in[45];
  const float* g2  = (const float*)d_in[46];
  const float* bb2 = (const float*)d_in[47];
  const float* W3  = (const float*)d_in[48];
  const float* b3  = (const float*)d_in[49];

  // ---- workspace carve (~380 MB) ----
  char* wsb = (char*)d_ws;
  size_t off = 0;
  auto carve = [&](size_t bytes) -> void* {
    void* p = wsb + off;
    off += (bytes + 255) & ~(size_t)255;
    return p;
  };
  _Float16* hf16  = (_Float16*)carve((size_t)N * HD * 2);
  _Float16* wf16  = (_Float16*)carve((size_t)HD * HD * 2);
  float*    xl    = (float*)carve((size_t)N * HD * 4);
  float*    xr    = (float*)carve((size_t)N * HD * 4);
  float*    acc   = (float*)carve((size_t)N * HD * 4);
  float*    logit = (float*)carve((size_t)E * HEADS * 4);
  float*    exv   = (float*)carve((size_t)E * HEADS * 4);
  unsigned* umax  = (unsigned*)carve((size_t)N * HEADS * 4);
  float*    den   = (float*)carve((size_t)N * HEADS * 4);
  float*    mu    = (float*)carve(HD * 4);
  float*    rs    = (float*)carve(HD * 4);
  float*    pooled= (float*)carve((size_t)G * HD * 4);
  float*    z1    = (float*)carve((size_t)G * 64 * 4);
  float*    z2    = (float*)carve((size_t)G * 64 * 4);
  float*    mu2   = (float*)carve(64 * 4);
  float*    rs2   = (float*)carve(64 * 4);
  (void)ws_size; (void)n_in;

  auto cdiv = [](long a, long b) -> int { return (int)((a + b - 1) / b); };
  const int GS = 2048;   // grid-stride block count for elementwise kernels

  // 1) embedding -> hf16 packed [N x 96]
  k_embed<<<cdiv(N, 256), 256, 0, stream>>>(x, emb[0], emb[1], emb[2], emb[3],
                                            emb[4], emb[5], emb[6], emb[7], hf16, N);

  // 2) GAT layers
  for (int l = 0; l < 4; ++l) {
    const int Ksrc = (l == 0) ? 80 : HD;
    const int K    = (l == 0) ? 96 : HD;
    dim3 gg(HD / 128, cdiv(N, 16));           // block tile 16 x 128

    k_b_pack<<<GS, 256, 0, stream>>>(L[l].Wl, wf16, Ksrc, K, HD);
    if (l == 0) k_gemm_f16_wmma<96><<<gg, 128, 0, stream>>>(hf16, wf16, xl, N, HD);
    else        k_gemm_f16_wmma<512><<<gg, 128, 0, stream>>>(hf16, wf16, xl, N, HD);

    k_b_pack<<<GS, 256, 0, stream>>>(L[l].Wr, wf16, Ksrc, K, HD);
    if (l == 0) k_gemm_f16_wmma<96><<<gg, 128, 0, stream>>>(hf16, wf16, xr, N, HD);
    else        k_gemm_f16_wmma<512><<<gg, 128, 0, stream>>>(hf16, wf16, xr, N, HD);

    k_fill_u32<<<GS, 256, 0, stream>>>(umax, 0u, (long)N * HEADS);   // enc(-inf)
    k_fill_f32<<<GS, 256, 0, stream>>>(den, 0.f, (long)N * HEADS);

    k_edge_logit<<<cdiv(E, 8), 256, 0, stream>>>(xl, xr, eattr, eidx, L[l].We,
                                                 L[l].att, logit, umax, E);
    k_edge_exp<<<GS, 256, 0, stream>>>(logit, eidx, umax, exv, den, E);

    k_init_bias<<<GS, 256, 0, stream>>>(acc, L[l].bias, (long)N * HD);
    k_edge_accum<<<cdiv(E, 8), 256, 0, stream>>>(xl, eidx, exv, den, acc, E);

    k_bn_stats<<<HD, 256, 0, stream>>>(acc, N, HD, mu, rs);
    k_bn_leaky<<<GS, 256, 0, stream>>>(acc, mu, rs, L[l].g, L[l].b, hf16,
                                       (long)N * HD);
  }

  // 3) global add pool
  k_fill_f32<<<GS, 256, 0, stream>>>(pooled, 0.f, (long)G * HD);
  k_pool<<<GS, 256, 0, stream>>>(acc, batch, pooled, (long)N * HD);

  // 4) MLP head
  k_lin<<<cdiv((long)G * 64, 256), 256, 0, stream>>>(pooled, W1, b1, z1, G, HD, 64);
  k_bn_stats<<<64, 256, 0, stream>>>(z1, G, 64, mu2, rs2);
  k_bn_relu<<<cdiv((long)G * 64, 256), 256, 0, stream>>>(z1, mu2, rs2, g1, bb1, G, 64);

  k_lin<<<cdiv((long)G * 64, 256), 256, 0, stream>>>(z1, W2, b2, z2, G, 64, 64);
  k_bn_stats<<<64, 256, 0, stream>>>(z2, G, 64, mu2, rs2);
  k_bn_relu<<<cdiv((long)G * 64, 256), 256, 0, stream>>>(z2, mu2, rs2, g2, bb2, G, 64);

  k_lin<<<cdiv((long)G * 2, 256), 256, 0, stream>>>(z2, W3, b3, (float*)d_out, G, 64, 2);
}
